// VectorProteinGNN_PocketMiner_18013092839589
// MI455X (gfx1250) — compile-verified, matrix operand order checked by reading, not run on hardware
//
#include <hip/hip_runtime.h>
#include <cstdint>
#include <cstddef>

// ---------------------------------------------------------------------------
// GVP-GNN forward for MI455X (gfx1250, wave32).
// Heavy GEMMs (edge-message Ws, node feed-forward Ws) run on WMMA f16->f32,
// with one A-fragment load per k-step shared across TG output column tiles
// (cuts dominant A-matrix HBM traffic by TG x). Small vector-channel einsums
// (Wh/Wv over 3D vectors), LayerNorms, gathers and the segment-sum
// aggregation run as f32 VALU kernels; aggregation is one atomic per thread.
// ---------------------------------------------------------------------------

typedef _Float16 f16_t;
typedef __attribute__((ext_vector_type(16))) _Float16 v16h;
typedef __attribute__((ext_vector_type(8)))  _Float16 v8h;
typedef __attribute__((ext_vector_type(8)))  float    v8f;

#define EPSN  1e-8f
#define LNEPS 1e-5f

constexpr int NN  = 25000;   // nodes
constexpr int EE  = 400000;  // edges (multiple of 16)
constexpr int NP  = 25008;   // nodes padded to multiple of 16 for WMMA tiles
constexpr int SLD = 112;     // leading dim of 100-wide scalar features (7 tiles)

// K-padded GEMM shapes (K must be a multiple of 32 for 16x16x32 WMMA steps)
constexpr int K_M0 = 288;    // msg gvp0: 232+33 = 265 -> 288
constexpr int K_M1 = 128;    // msg gvp1/2: 100+16 = 116 -> 128
constexpr int K_F0 = 160;    // ff0: 100+32 = 132 -> 160
constexpr int K_F1 = 448;    // ff1: 400+32 = 432 -> 448

// ---------------------------------------------------------------------------
// device helpers
// ---------------------------------------------------------------------------
__device__ inline float vnrm(float a, float b, float c) {
  return sqrtf(fmaxf(a * a + b * b + c * c, EPSN));
}

__device__ inline void ln_s(float* s, int n, const float* g, const float* b) {
  float mu = 0.f;
  for (int k = 0; k < n; ++k) mu += s[k];
  mu /= (float)n;
  float var = 0.f;
  for (int k = 0; k < n; ++k) { float d = s[k] - mu; var += d * d; }
  var /= (float)n;
  float inv = rsqrtf(var + LNEPS);
  for (int k = 0; k < n; ++k) s[k] = (s[k] - mu) * inv * g[k] + b[k];
}

__device__ inline void ln_v(float (*v)[3], int nv) {
  float ms = 0.f;
  for (int h = 0; h < nv; ++h) {
    float d = v[h][0]*v[h][0] + v[h][1]*v[h][1] + v[h][2]*v[h][2];
    ms += fmaxf(d, EPSN);
  }
  ms /= (float)nv;
  float inv = rsqrtf(ms);
  for (int h = 0; h < nv; ++h) { v[h][0]*=inv; v[h][1]*=inv; v[h][2]*=inv; }
}

// ---------------------------------------------------------------------------
// WMMA GEMM:  C[rows x nvalid] = A[rows x K] * W^T + bias
//   A: f16 row-major, K multiple of 32, rows multiple of 16
//   W: f16 row-major [ncols_pad x K] (i.e. B^T); bias: f32 [nvalid]
// One wave per (16-row tile x TG 16-col tiles). The A fragment is loaded once
// per k-step and reused across all TG B tiles (TG f32 accumulator sets).
// CDNA5 16x16x32 f16 fragment layout per lane l (row r = l&15):
//   halves[0:8]  = row[k0 + base .. +7]
//   halves[8:16] = row[k0 + base + 16 .. +23],  base = (l>>4)*8
// ---------------------------------------------------------------------------
template <int TG>
__global__ __launch_bounds__(32)
void wmma_gemm_kernel(const f16_t* __restrict__ A, const f16_t* __restrict__ W,
                      const float* __restrict__ bias, float* __restrict__ C,
                      int K, int ldC, int nvalid)
{
  int lane = threadIdx.x;
  int rt = blockIdx.x;            // 16-row tile
  int cg = blockIdx.y;            // group of TG 16-col tiles
  int r  = lane & 15;
  int hi = lane >> 4;             // 0 or 1
  const f16_t* arow = A + (size_t)(rt * 16 + r) * K + hi * 8;
  const f16_t* wrow = W + (size_t)(cg * TG * 16 + r) * K + hi * 8;
  v8f acc[TG];
#pragma unroll
  for (int t = 0; t < TG; ++t) acc[t] = {};
  for (int k0 = 0; k0 < K; k0 += 32) {
    // A is streamed exactly once per wave: prefetch ahead of the pipeline
    __builtin_prefetch((const void*)(arow + k0 + 256), 0, 1);
    v8h alo = *(const v8h*)(arow + k0);
    v8h ahi = *(const v8h*)(arow + k0 + 16);
    v16h a = __builtin_shufflevector(alo, ahi, 0,1,2,3,4,5,6,7,8,9,10,11,12,13,14,15);
#pragma unroll
    for (int t = 0; t < TG; ++t) {
      const f16_t* wr = wrow + (size_t)t * 16 * K;
      v8h blo = *(const v8h*)(wr + k0);
      v8h bhi = *(const v8h*)(wr + k0 + 16);
      v16h b = __builtin_shufflevector(blo, bhi, 0,1,2,3,4,5,6,7,8,9,10,11,12,13,14,15);
      acc[t] = __builtin_amdgcn_wmma_f32_16x16x32_f16(false, a, false, b,
                                                      (short)0, acc[t], false, false);
    }
  }
  int mbase = rt * 16 + hi * 8;
#pragma unroll
  for (int t = 0; t < TG; ++t) {
    int n = (cg * TG + t) * 16 + r;
    if (n < nvalid) {
      float bv = bias[n];
#pragma unroll
      for (int i = 0; i < 8; ++i)
        C[(size_t)(mbase + i) * ldC + n] = acc[t][i] + bv;
    }
  }
}

// ---------------------------------------------------------------------------
// weight pack: f32 [n x k] -> f16 [npad x kpad] zero-padded row-major
// ---------------------------------------------------------------------------
__global__ void pack_w_kernel(const float* __restrict__ src, f16_t* __restrict__ dst,
                              int n, int k, int kpad, int npad)
{
  int idx = blockIdx.x * blockDim.x + threadIdx.x;
  int tot = npad * kpad;
  if (idx >= tot) return;
  int r = idx / kpad, c = idx - r * kpad;
  float v = (r < n && c < k) ? src[r * k + c] : 0.f;
  dst[idx] = (f16_t)v;
}

__global__ void zero_f32_kernel(float* __restrict__ p, long n)
{
  long i = (long)blockIdx.x * blockDim.x + threadIdx.x;
  if (i < n) p[i] = 0.f;
}

// ---------------------------------------------------------------------------
// node init: node_proj(6,3->64,8) + LN + ntype-emb concat + LN + gvp_node(80,8->100,16)
// ---------------------------------------------------------------------------
__global__ __launch_bounds__(128)
void node_init_kernel(const float* __restrict__ x_s, const float* __restrict__ x_v,
                      const int* __restrict__ ntypes, const float* __restrict__ ntype_emb,
                      const float* npWh, const float* npWs, const float* npWv, const float* npbs,
                      const float* nplnG, const float* nplnB,
                      const float* gnlnG, const float* gnlnB,
                      const float* gnWh, const float* gnWs, const float* gnWv, const float* gnbs,
                      float* __restrict__ s_node, float* __restrict__ v_node)
{
  int i = blockIdx.x * blockDim.x + threadIdx.x;
  if (i >= NN) return;
  float xv[3][3];
  for (int r = 0; r < 3; ++r)
    for (int c = 0; c < 3; ++c) xv[r][c] = x_v[i * 9 + r * 3 + c];
  float vh8[8][3];
  for (int h = 0; h < 8; ++h)
    for (int c = 0; c < 3; ++c) {
      float a = 0.f;
      for (int j = 0; j < 3; ++j) a += npWh[h * 3 + j] * xv[j][c];
      vh8[h][c] = a;
    }
  float in14[14];
  for (int k = 0; k < 6; ++k) in14[k] = x_s[i * 6 + k];
  for (int h = 0; h < 8; ++h) in14[6 + h] = vnrm(vh8[h][0], vh8[h][1], vh8[h][2]);
  float s64[64];
  for (int o = 0; o < 64; ++o) {
    float a = npbs[o];
    for (int k = 0; k < 14; ++k) a += npWs[o * 14 + k] * in14[k];
    s64[o] = a;
  }
  float v8[8][3];
  for (int o = 0; o < 8; ++o)
    for (int c = 0; c < 3; ++c) {
      float a = 0.f;
      for (int h = 0; h < 8; ++h) a += npWv[o * 8 + h] * vh8[h][c];
      v8[o][c] = a;
    }
  ln_s(s64, 64, nplnG, nplnB);
  ln_v(v8, 8);
  float s80[80];
  int nt = ntypes[i];
  for (int k = 0; k < 16; ++k) s80[k] = ntype_emb[nt * 16 + k];
  for (int k = 0; k < 64; ++k) s80[16 + k] = s64[k];
  ln_s(s80, 80, gnlnG, gnlnB);
  ln_v(v8, 8);
  float in96[96];
  for (int k = 0; k < 80; ++k) in96[k] = s80[k];
  float v16[16][3];
  for (int o = 0; o < 16; ++o) { v16[o][0]=0.f; v16[o][1]=0.f; v16[o][2]=0.f; }
  for (int h = 0; h < 16; ++h) {
    float a=0.f, b=0.f, c=0.f;
    for (int j = 0; j < 8; ++j) {
      float w = gnWh[h * 8 + j];
      a += w * v8[j][0]; b += w * v8[j][1]; c += w * v8[j][2];
    }
    in96[80 + h] = vnrm(a, b, c);
    for (int o = 0; o < 16; ++o) {
      float w = gnWv[o * 16 + h];
      v16[o][0] += w * a; v16[o][1] += w * b; v16[o][2] += w * c;
    }
  }
  for (int o = 0; o < 100; ++o) {
    float a = gnbs[o];
    for (int k = 0; k < 96; ++k) a += gnWs[o * 96 + k] * in96[k];
    s_node[(size_t)i * SLD + o] = a;
  }
  for (int o = 0; o < 16; ++o)
    for (int c = 0; c < 3; ++c) v_node[(size_t)i * 48 + o * 3 + c] = v16[o][c];
}

// ---------------------------------------------------------------------------
// edge init: edge_proj(32,1->32,1)+LN + etype-emb + LN + gvp_edge(48,1->32,1)
// ---------------------------------------------------------------------------
__global__ __launch_bounds__(256)
void edge_init_kernel(const float* __restrict__ eattr_s, const float* __restrict__ eattr_v,
                      const int* __restrict__ etypes, const float* __restrict__ etype_emb,
                      const float* epWh, const float* epWs, const float* epWv, const float* epbs,
                      const float* eplnG, const float* eplnB,
                      const float* gelnG, const float* gelnB,
                      const float* geWh, const float* geWs, const float* geWv, const float* gebs,
                      float* __restrict__ es_out, float* __restrict__ ev_out)
{
  int e = blockIdx.x * blockDim.x + threadIdx.x;
  if (e >= EE) return;
  float wh0 = epWh[0];
  float vh0[1][3];
  for (int c = 0; c < 3; ++c) vh0[0][c] = wh0 * eattr_v[e * 3 + c];
  float in33[33];
  for (int k = 0; k < 32; ++k) in33[k] = eattr_s[e * 32 + k];
  in33[32] = vnrm(vh0[0][0], vh0[0][1], vh0[0][2]);
  float s32[32];
  for (int o = 0; o < 32; ++o) {
    float a = epbs[o];
    for (int k = 0; k < 33; ++k) a += epWs[o * 33 + k] * in33[k];
    s32[o] = a;
  }
  float v1[1][3];
  float wv0 = epWv[0];
  for (int c = 0; c < 3; ++c) v1[0][c] = wv0 * vh0[0][c];
  ln_s(s32, 32, eplnG, eplnB);
  ln_v(v1, 1);
  float s48[48];
  int et = etypes[e];
  for (int k = 0; k < 16; ++k) s48[k] = etype_emb[et * 16 + k];
  for (int k = 0; k < 32; ++k) s48[16 + k] = s32[k];
  ln_s(s48, 48, gelnG, gelnB);
  ln_v(v1, 1);
  float wh1 = geWh[0];
  float vh1[3];
  for (int c = 0; c < 3; ++c) vh1[c] = wh1 * v1[0][c];
  float in49[49];
  for (int k = 0; k < 48; ++k) in49[k] = s48[k];
  in49[48] = vnrm(vh1[0], vh1[1], vh1[2]);
  for (int o = 0; o < 32; ++o) {
    float a = gebs[o];
    for (int k = 0; k < 49; ++k) a += geWs[o * 49 + k] * in49[k];
    es_out[(size_t)e * 32 + o] = a;
  }
  float wv1 = geWv[0];
  for (int c = 0; c < 3; ++c) ev_out[(size_t)e * 3 + c] = wv1 * vh1[c];
}

// ---------------------------------------------------------------------------
// msg GVP #0 build: gather [s_src|es|s_dst|vn(Wh@[v_src;ev;v_dst])] -> f16 A row,
// and mv1 = Wv @ vh
// ---------------------------------------------------------------------------
__global__ __launch_bounds__(128)
void msg0_build_kernel(const int* __restrict__ ei,
                       const float* __restrict__ s_node, const float* __restrict__ v_node,
                       const float* __restrict__ es, const float* __restrict__ ev,
                       const float* __restrict__ Wh, const float* __restrict__ Wv,
                       f16_t* __restrict__ A, float* __restrict__ mv_out)
{
  int e = blockIdx.x * blockDim.x + threadIdx.x;
  if (e >= EE) return;
  int src = ei[e], dst = ei[EE + e];
  f16_t* arow = A + (size_t)e * K_M0;
  for (int k = 0; k < 100; ++k) arow[k]       = (f16_t)s_node[(size_t)src * SLD + k];
  for (int k = 0; k < 32;  ++k) arow[100 + k] = (f16_t)es[(size_t)e * 32 + k];
  for (int k = 0; k < 100; ++k) arow[132 + k] = (f16_t)s_node[(size_t)dst * SLD + k];
  float mvin[33][3];
  for (int h = 0; h < 16; ++h)
    for (int c = 0; c < 3; ++c) mvin[h][c] = v_node[(size_t)src * 48 + h * 3 + c];
  for (int c = 0; c < 3; ++c) mvin[16][c] = ev[(size_t)e * 3 + c];
  for (int h = 0; h < 16; ++h)
    for (int c = 0; c < 3; ++c) mvin[17 + h][c] = v_node[(size_t)dst * 48 + h * 3 + c];
  float mo[16][3];
  for (int o = 0; o < 16; ++o) { mo[o][0]=0.f; mo[o][1]=0.f; mo[o][2]=0.f; }
  for (int h = 0; h < 33; ++h) {
    float a=0.f, b=0.f, c=0.f;
    for (int j = 0; j < 33; ++j) {
      float w = Wh[h * 33 + j];
      a += w * mvin[j][0]; b += w * mvin[j][1]; c += w * mvin[j][2];
    }
    arow[232 + h] = (f16_t)vnrm(a, b, c);
    for (int o = 0; o < 16; ++o) {
      float w = Wv[o * 33 + h];
      mo[o][0] += w * a; mo[o][1] += w * b; mo[o][2] += w * c;
    }
  }
  for (int k = 265; k < K_M0; ++k) arow[k] = (f16_t)0.f;
  for (int o = 0; o < 16; ++o)
    for (int c = 0; c < 3; ++c) mv_out[(size_t)e * 48 + o * 3 + c] = mo[o][c];
}

// ---------------------------------------------------------------------------
// msg GVP #1/#2 build (100,16 -> 100,16): A = [ms_prev|vn], mv_out = Wv @ vh
// ---------------------------------------------------------------------------
__global__ __launch_bounds__(128)
void msg_mid_build_kernel(const float* __restrict__ C, const float* __restrict__ mv_in,
                          const float* __restrict__ Wh, const float* __restrict__ Wv,
                          f16_t* __restrict__ A, float* __restrict__ mv_out)
{
  int e = blockIdx.x * blockDim.x + threadIdx.x;
  if (e >= EE) return;
  f16_t* arow = A + (size_t)e * K_M1;
  for (int k = 0; k < 100; ++k) arow[k] = (f16_t)C[(size_t)e * SLD + k];
  float mvp[16][3];
  for (int h = 0; h < 16; ++h)
    for (int c = 0; c < 3; ++c) mvp[h][c] = mv_in[(size_t)e * 48 + h * 3 + c];
  float mo[16][3];
  for (int o = 0; o < 16; ++o) { mo[o][0]=0.f; mo[o][1]=0.f; mo[o][2]=0.f; }
  for (int h = 0; h < 16; ++h) {
    float a=0.f, b=0.f, c=0.f;
    for (int j = 0; j < 16; ++j) {
      float w = Wh[h * 16 + j];
      a += w * mvp[j][0]; b += w * mvp[j][1]; c += w * mvp[j][2];
    }
    arow[100 + h] = (f16_t)vnrm(a, b, c);
    for (int o = 0; o < 16; ++o) {
      float w = Wv[o * 16 + h];
      mo[o][0] += w * a; mo[o][1] += w * b; mo[o][2] += w * c;
    }
  }
  for (int k = 116; k < K_M1; ++k) arow[k] = (f16_t)0.f;
  for (int o = 0; o < 16; ++o)
    for (int c = 0; c < 3; ++c) mv_out[(size_t)e * 48 + o * 3 + c] = mo[o][c];
}

// ---------------------------------------------------------------------------
// segment-sum aggregation over dst: one f32 global atomic per thread, with
// consecutive threads on consecutive channels (coalesced atomic bursts)
// ---------------------------------------------------------------------------
__global__ __launch_bounds__(256)
void aggregate_s_kernel(const int* __restrict__ ei, const float* __restrict__ C,
                        float* __restrict__ dhs)
{
  long t = (long)blockIdx.x * blockDim.x + threadIdx.x;
  if (t >= (long)EE * 100) return;
  int e = (int)(t / 100);
  int k = (int)(t - (long)e * 100);
  int dst = ei[EE + e];
  atomicAdd(&dhs[(size_t)dst * 100 + k], C[(size_t)e * SLD + k]);
}

__global__ __launch_bounds__(256)
void aggregate_v_kernel(const int* __restrict__ ei, const float* __restrict__ mv,
                        float* __restrict__ dhv)
{
  long t = (long)blockIdx.x * blockDim.x + threadIdx.x;
  if (t >= (long)EE * 48) return;
  int e = (int)(t / 48);
  int k = (int)(t - (long)e * 48);
  int dst = ei[EE + e];
  atomicAdd(&dhv[(size_t)dst * 48 + k], mv[(size_t)e * 48 + k]);
}

__global__ __launch_bounds__(256)
void aggregate_cnt_kernel(const int* __restrict__ ei, float* __restrict__ cnt)
{
  int e = blockIdx.x * blockDim.x + threadIdx.x;
  if (e >= EE) return;
  atomicAdd(&cnt[ei[EE + e]], 1.f);
}

// ---------------------------------------------------------------------------
// node update stage A: residual + LN0, then build ff0 A-row and hv0 = Wv0 @ vh
// ---------------------------------------------------------------------------
__global__ __launch_bounds__(128)
void node_update_a_kernel(const float* __restrict__ s_node, const float* __restrict__ v_node,
                          const float* __restrict__ dhs, const float* __restrict__ dhv,
                          const float* __restrict__ cnt,
                          const float* ln0g, const float* ln0b,
                          const float* f0Wh, const float* f0Wv,
                          float* __restrict__ s_mid, float* __restrict__ v_mid,
                          f16_t* __restrict__ A, float* __restrict__ hv0)
{
  int i = blockIdx.x * blockDim.x + threadIdx.x;
  if (i >= NP) return;
  f16_t* arow = A + (size_t)i * K_F0;
  if (i >= NN) {  // zero the pad rows so the WMMA GEMM reads clean data
    for (int k = 0; k < K_F0; ++k) arow[k] = (f16_t)0.f;
    return;
  }
  float cinv = 1.f / fmaxf(cnt[i], 1.f);
  float s[100];
  for (int k = 0; k < 100; ++k) s[k] = s_node[(size_t)i * SLD + k] + dhs[(size_t)i * 100 + k] * cinv;
  float v[16][3];
  for (int h = 0; h < 16; ++h)
    for (int c = 0; c < 3; ++c)
      v[h][c] = v_node[(size_t)i * 48 + h * 3 + c] + dhv[(size_t)i * 48 + h * 3 + c] * cinv;
  ln_s(s, 100, ln0g, ln0b);
  ln_v(v, 16);
  for (int k = 0; k < 100; ++k) { s_mid[(size_t)i * 100 + k] = s[k]; arow[k] = (f16_t)s[k]; }
  for (int h = 0; h < 16; ++h)
    for (int c = 0; c < 3; ++c) v_mid[(size_t)i * 48 + h * 3 + c] = v[h][c];
  float hv[32][3];
  for (int o = 0; o < 32; ++o) { hv[o][0]=0.f; hv[o][1]=0.f; hv[o][2]=0.f; }
  for (int h = 0; h < 32; ++h) {
    float a=0.f, b=0.f, c=0.f;
    for (int j = 0; j < 16; ++j) {
      float w = f0Wh[h * 16 + j];
      a += w * v[j][0]; b += w * v[j][1]; c += w * v[j][2];
    }
    arow[100 + h] = (f16_t)vnrm(a, b, c);
    for (int o = 0; o < 32; ++o) {
      float w = f0Wv[o * 32 + h];
      hv[o][0] += w * a; hv[o][1] += w * b; hv[o][2] += w * c;
    }
  }
  for (int k = 132; k < K_F0; ++k) arow[k] = (f16_t)0.f;
  for (int o = 0; o < 32; ++o)
    for (int c = 0; c < 3; ++c) hv0[(size_t)i * 96 + o * 3 + c] = hv[o][c];
}

// ---------------------------------------------------------------------------
// node update stage B: build ff1 A-row from hs(400)+vn2, hv2 = Wv1 @ vh2
// ---------------------------------------------------------------------------
__global__ __launch_bounds__(128)
void node_update_b_kernel(const float* __restrict__ Cff0, const float* __restrict__ hv0,
                          const float* f1Wh, const float* f1Wv,
                          f16_t* __restrict__ A, float* __restrict__ hv2)
{
  int i = blockIdx.x * blockDim.x + threadIdx.x;
  if (i >= NP) return;
  f16_t* arow = A + (size_t)i * K_F1;
  if (i >= NN) {
    for (int k = 0; k < K_F1; ++k) arow[k] = (f16_t)0.f;
    return;
  }
  for (int k = 0; k < 400; ++k) arow[k] = (f16_t)Cff0[(size_t)i * 400 + k];
  float hvp[32][3];
  for (int h = 0; h < 32; ++h)
    for (int c = 0; c < 3; ++c) hvp[h][c] = hv0[(size_t)i * 96 + h * 3 + c];
  float ho[16][3];
  for (int o = 0; o < 16; ++o) { ho[o][0]=0.f; ho[o][1]=0.f; ho[o][2]=0.f; }
  for (int h = 0; h < 32; ++h) {
    float a=0.f, b=0.f, c=0.f;
    for (int j = 0; j < 32; ++j) {
      float w = f1Wh[h * 32 + j];
      a += w * hvp[j][0]; b += w * hvp[j][1]; c += w * hvp[j][2];
    }
    arow[400 + h] = (f16_t)vnrm(a, b, c);
    for (int o = 0; o < 16; ++o) {
      float w = f1Wv[o * 32 + h];
      ho[o][0] += w * a; ho[o][1] += w * b; ho[o][2] += w * c;
    }
  }
  for (int k = 432; k < K_F1; ++k) arow[k] = (f16_t)0.f;
  for (int o = 0; o < 16; ++o)
    for (int c = 0; c < 3; ++c) hv2[(size_t)i * 48 + o * 3 + c] = ho[o][c];
}

// ---------------------------------------------------------------------------
// node update stage C: residual + LN1 -> write back s_node / v_node
// ---------------------------------------------------------------------------
__global__ __launch_bounds__(128)
void node_update_c_kernel(const float* __restrict__ Cff1, const float* __restrict__ hv2,
                          const float* __restrict__ s_mid, const float* __restrict__ v_mid,
                          const float* ln1g, const float* ln1b,
                          float* __restrict__ s_node, float* __restrict__ v_node)
{
  int i = blockIdx.x * blockDim.x + threadIdx.x;
  if (i >= NN) return;
  float s[100];
  for (int k = 0; k < 100; ++k) s[k] = s_mid[(size_t)i * 100 + k] + Cff1[(size_t)i * SLD + k];
  float v[16][3];
  for (int h = 0; h < 16; ++h)
    for (int c = 0; c < 3; ++c)
      v[h][c] = v_mid[(size_t)i * 48 + h * 3 + c] + hv2[(size_t)i * 48 + h * 3 + c];
  ln_s(s, 100, ln1g, ln1b);
  ln_v(v, 16);
  for (int k = 0; k < 100; ++k) s_node[(size_t)i * SLD + k] = s[k];
  for (int h = 0; h < 16; ++h)
    for (int c = 0; c < 3; ++c) v_node[(size_t)i * 48 + h * 3 + c] = v[h][c];
}

// ---------------------------------------------------------------------------
// final LN + to_scalar (100,16 -> 8 scalars)
// ---------------------------------------------------------------------------
__global__ __launch_bounds__(128)
void final_kernel(const float* __restrict__ s_node, const float* __restrict__ v_node,
                  const float* flng, const float* flnb,
                  const float* tsWh, const float* tsWs, const float* tsbs,
                  float* __restrict__ out)
{
  int i = blockIdx.x * blockDim.x + threadIdx.x;
  if (i >= NN) return;
  float s[100];
  for (int k = 0; k < 100; ++k) s[k] = s_node[(size_t)i * SLD + k];
  float v[16][3];
  for (int h = 0; h < 16; ++h)
    for (int c = 0; c < 3; ++c) v[h][c] = v_node[(size_t)i * 48 + h * 3 + c];
  ln_s(s, 100, flng, flnb);
  ln_v(v, 16);
  float vn16[16];
  for (int h = 0; h < 16; ++h) {
    float a=0.f, b=0.f, c=0.f;
    for (int j = 0; j < 16; ++j) {
      float w = tsWh[h * 16 + j];
      a += w * v[j][0]; b += w * v[j][1]; c += w * v[j][2];
    }
    vn16[h] = vnrm(a, b, c);
  }
  for (int o = 0; o < 8; ++o) {
    float a = tsbs[o];
    for (int k = 0; k < 100; ++k) a += tsWs[o * 116 + k] * s[k];
    for (int h = 0; h < 16; ++h)  a += tsWs[o * 116 + 100 + h] * vn16[h];
    out[(size_t)i * 8 + o] = a;
  }
}

// ---------------------------------------------------------------------------
// host launcher
// ---------------------------------------------------------------------------
extern "C" void kernel_launch(void* const* d_in, const int* in_sizes, int n_in,
                              void* d_out, int out_size, void* d_ws, size_t ws_size,
                              hipStream_t stream)
{
  (void)n_in; (void)out_size; (void)ws_size;
  // -------- input index decode (top-level insertion order, params JAX-sorted;
  //          fallback: fully sorted top level) --------
  int i_xs, i_xv, i_eas, i_eav, i_ei, i_nt, i_et, pbase;
  if (in_sizes[0] == NN * 6) {
    i_xs = 0; i_xv = 1; i_eas = 2; i_eav = 3; pbase = 4; i_ei = 83; i_nt = 84; i_et = 85;
  } else {
    i_eas = 0; i_eav = 1; i_ei = 2; i_et = 3; i_nt = 4; pbase = 5; i_xs = 84; i_xv = 85;
  }
  auto F = [&](int idx) -> const float* { return (const float*)d_in[idx]; };
  auto P = [&](int off) -> const float* { return (const float*)d_in[pbase + off]; };
  const float* x_s     = F(i_xs);
  const float* x_v     = F(i_xv);
  const float* eattr_s = F(i_eas);
  const float* eattr_v = F(i_eav);
  const int*   ei      = (const int*)d_in[i_ei];
  const int*   ntypes  = (const int*)d_in[i_nt];
  const int*   etypes  = (const int*)d_in[i_et];

  // params leaf layout (sorted keys; 'W' < 'b' in ASCII):
  // convs[l] (24 leaves each): ff0{Wh,Ws,Wv,bs} ff1{...} ln0{b,g} ln1{b,g}
  //                            msg0{Wh,Ws,Wv,bs} msg1{...} msg2{...}
  struct Conv {
    const float *f0Wh,*f0Ws,*f0Wv,*f0bs, *f1Wh,*f1Ws,*f1Wv,*f1bs;
    const float *ln0b,*ln0g,*ln1b,*ln1g;
    const float *m0Wh,*m0Ws,*m0Wv,*m0bs, *m1Wh,*m1Ws,*m1Wv,*m1bs, *m2Wh,*m2Ws,*m2Wv,*m2bs;
  };
  Conv cv[2];
  for (int l = 0; l < 2; ++l) {
    int b = 24 * l;
    cv[l].f0Wh = P(b+0);  cv[l].f0Ws = P(b+1);  cv[l].f0Wv = P(b+2);  cv[l].f0bs = P(b+3);
    cv[l].f1Wh = P(b+4);  cv[l].f1Ws = P(b+5);  cv[l].f1Wv = P(b+6);  cv[l].f1bs = P(b+7);
    cv[l].ln0b = P(b+8);  cv[l].ln0g = P(b+9);  cv[l].ln1b = P(b+10); cv[l].ln1g = P(b+11);
    cv[l].m0Wh = P(b+12); cv[l].m0Ws = P(b+13); cv[l].m0Wv = P(b+14); cv[l].m0bs = P(b+15);
    cv[l].m1Wh = P(b+16); cv[l].m1Ws = P(b+17); cv[l].m1Wv = P(b+18); cv[l].m1bs = P(b+19);
    cv[l].m2Wh = P(b+20); cv[l].m2Ws = P(b+21); cv[l].m2Wv = P(b+22); cv[l].m2bs = P(b+23);
  }
  const float* epWh = P(48);  const float* epWs = P(49);
  const float* epWv = P(50);  const float* epbs = P(51);
  const float* eplnB = P(52); const float* eplnG = P(53);
  const float* etype_emb = P(54);
  const float* flnB = P(55);  const float* flnG = P(56);
  const float* geWh = P(57);  const float* geWs = P(58);
  const float* geWv = P(59);  const float* gebs = P(60);
  const float* gelnB = P(61); const float* gelnG = P(62);
  const float* gnWh = P(63);  const float* gnWs = P(64);
  const float* gnWv = P(65);  const float* gnbs = P(66);
  const float* gnlnB = P(67); const float* gnlnG = P(68);
  const float* npWh = P(69);  const float* npWs = P(70);
  const float* npWv = P(71);  const float* npbs = P(72);
  const float* nplnB = P(73); const float* nplnG = P(74);
  const float* ntype_emb = P(75);
  const float* tsWh = P(76);  const float* tsWs = P(77);  const float* tsbs = P(78);

  // -------- workspace carve --------
  size_t off = 0;
  auto carve = [&](size_t bytes) -> char* {
    char* p = (char*)d_ws + off;
    off = (off + bytes + 255) & ~(size_t)255;
    return p;
  };
  f16_t* A      = (f16_t*)carve((size_t)EE * K_M0 * 2);  // also reused for node A (NP*K_F1 fits)
  float* C      = (float*)carve((size_t)EE * SLD * 4);   // also reused for ff0 output (NP*400 fits)
  float* s_node = (float*)carve((size_t)NP * SLD * 4);
  float* v_node = (float*)carve((size_t)NN * 48 * 4);
  float* es     = (float*)carve((size_t)EE * 32 * 4);
  float* ev     = (float*)carve((size_t)EE * 3 * 4);
  float* mvA    = (float*)carve((size_t)EE * 48 * 4);    // reused for hv2 (NN*48)
  float* mvB    = (float*)carve((size_t)EE * 48 * 4);    // reused for hv0 (NN*96)
  float* dhs    = (float*)carve((size_t)NN * 100 * 4);
  float* dhv    = (float*)carve((size_t)NN * 48 * 4);
  float* cnt    = (float*)carve((size_t)NN * 4);
  float* s_mid  = (float*)carve((size_t)NN * 100 * 4);
  float* v_mid  = (float*)carve((size_t)NN * 48 * 4);
  f16_t* WpM0[2]; f16_t* WpM1[2]; f16_t* WpM2[2]; f16_t* WpF0[2]; f16_t* WpF1[2];
  for (int l = 0; l < 2; ++l) {
    WpM0[l] = (f16_t*)carve((size_t)112 * K_M0 * 2);
    WpM1[l] = (f16_t*)carve((size_t)112 * K_M1 * 2);
    WpM2[l] = (f16_t*)carve((size_t)112 * K_M1 * 2);
    WpF0[l] = (f16_t*)carve((size_t)400 * K_F0 * 2);
    WpF1[l] = (f16_t*)carve((size_t)112 * K_F1 * 2);
  }

  auto pack = [&](const float* src, f16_t* dst, int n, int k, int kpad, int npad) {
    int tot = npad * kpad;
    pack_w_kernel<<<(tot + 255) / 256, 256, 0, stream>>>(src, dst, n, k, kpad, npad);
  };
  auto zero = [&](float* p, long n) {
    zero_f32_kernel<<<(unsigned)((n + 255) / 256), 256, 0, stream>>>(p, n);
  };
  // 112-col GEMMs: one wave covers all 7 column tiles (A loaded once).
  auto gemm7 = [&](const f16_t* a, const f16_t* w, const float* bias, float* c,
                   int rows, int K, int ldC, int nvalid) {
    wmma_gemm_kernel<7><<<dim3(rows / 16, 1), 32, 0, stream>>>(a, w, bias, c, K, ldC, nvalid);
  };
  // 400-col GEMM: 5 groups of 5 column tiles.
  auto gemm25 = [&](const f16_t* a, const f16_t* w, const float* bias, float* c,
                    int rows, int K, int ldC, int nvalid) {
    wmma_gemm_kernel<5><<<dim3(rows / 16, 5), 32, 0, stream>>>(a, w, bias, c, K, ldC, nvalid);
  };

  // -------- pack all GEMM weights to padded f16 --------
  for (int l = 0; l < 2; ++l) {
    pack(cv[l].m0Ws, WpM0[l], 100, 265, K_M0, 112);
    pack(cv[l].m1Ws, WpM1[l], 100, 116, K_M1, 112);
    pack(cv[l].m2Ws, WpM2[l], 100, 116, K_M1, 112);
    pack(cv[l].f0Ws, WpF0[l], 400, 132, K_F0, 400);
    pack(cv[l].f1Ws, WpF1[l], 100, 432, K_F1, 112);
  }

  // -------- embeddings / projections --------
  node_init_kernel<<<(NN + 127) / 128, 128, 0, stream>>>(
      x_s, x_v, ntypes, ntype_emb,
      npWh, npWs, npWv, npbs, nplnG, nplnB,
      gnlnG, gnlnB, gnWh, gnWs, gnWv, gnbs,
      s_node, v_node);
  edge_init_kernel<<<(EE + 255) / 256, 256, 0, stream>>>(
      eattr_s, eattr_v, etypes, etype_emb,
      epWh, epWs, epWv, epbs, eplnG, eplnB,
      gelnG, gelnB, geWh, geWs, geWv, gebs,
      es, ev);

  // -------- conv layers --------
  for (int l = 0; l < 2; ++l) {
    msg0_build_kernel<<<(EE + 127) / 128, 128, 0, stream>>>(
        ei, s_node, v_node, es, ev, cv[l].m0Wh, cv[l].m0Wv, A, mvA);
    gemm7(A, WpM0[l], cv[l].m0bs, C, EE, K_M0, SLD, 100);

    msg_mid_build_kernel<<<(EE + 127) / 128, 128, 0, stream>>>(
        C, mvA, cv[l].m1Wh, cv[l].m1Wv, A, mvB);
    gemm7(A, WpM1[l], cv[l].m1bs, C, EE, K_M1, SLD, 100);

    msg_mid_build_kernel<<<(EE + 127) / 128, 128, 0, stream>>>(
        C, mvB, cv[l].m2Wh, cv[l].m2Wv, A, mvA);
    gemm7(A, WpM2[l], cv[l].m2bs, C, EE, K_M1, SLD, 100);

    zero(dhs, (long)NN * 100);
    zero(dhv, (long)NN * 48);
    zero(cnt, (long)NN);
    {
      long ts = (long)EE * 100;
      aggregate_s_kernel<<<(unsigned)((ts + 255) / 256), 256, 0, stream>>>(ei, C, dhs);
      long tv = (long)EE * 48;
      aggregate_v_kernel<<<(unsigned)((tv + 255) / 256), 256, 0, stream>>>(ei, mvA, dhv);
      aggregate_cnt_kernel<<<(EE + 255) / 256, 256, 0, stream>>>(ei, cnt);
    }

    float* hv0 = mvB;  // NN*96 floats, fits
    float* hv2 = mvA;  // NN*48 floats, fits
    node_update_a_kernel<<<(NP + 127) / 128, 128, 0, stream>>>(
        s_node, v_node, dhs, dhv, cnt, cv[l].ln0g, cv[l].ln0b,
        cv[l].f0Wh, cv[l].f0Wv, s_mid, v_mid, A, hv0);
    gemm25(A, WpF0[l], cv[l].f0bs, C, NP, K_F0, 400, 400);

    node_update_b_kernel<<<(NP + 127) / 128, 128, 0, stream>>>(
        C, hv0, cv[l].f1Wh, cv[l].f1Wv, A, hv2);
    gemm7(A, WpF1[l], cv[l].f1bs, C, NP, K_F1, SLD, 100);

    node_update_c_kernel<<<(NN + 127) / 128, 128, 0, stream>>>(
        C, hv2, s_mid, v_mid, cv[l].ln1g, cv[l].ln1b, s_node, v_node);
  }

  // -------- readout --------
  final_kernel<<<(NN + 127) / 128, 128, 0, stream>>>(
      s_node, v_node, flnG, flnB, tsWh, tsWs, tsbs, (float*)d_out);
}
